// DisNet_73289321939440
// MI455X (gfx1250) — matrix operand order repaired and checked by reading, not compiled, not run on hardware
//
#include <hip/hip_runtime.h>
#include <hip/hip_bf16.h>

typedef unsigned int        u32;
typedef unsigned long long  u64;
typedef unsigned short      u16;

typedef __attribute__((ext_vector_type(16))) __bf16 v16bf;
typedef __attribute__((ext_vector_type(8)))  float  v8f;

#define NS    4096
#define NTOT  8192
#define DIM   256
#define KSPL  512      // hi|lo split K
#define BM    128
#define BN    128
#define BK    32
#define LDA   40       // padded LDS row stride (u16 units) to dodge bank conflicts

__device__ __forceinline__ u16 f32_to_bf16(float f) {
    u32 u = __float_as_uint(f);
    u32 r = u + 0x7FFFu + ((u >> 16) & 1u);     // round-to-nearest-even
    return (u16)(r >> 16);
}
__device__ __forceinline__ float bf16_to_f32(u16 h) {
    return __uint_as_float(((u32)h) << 16);
}

// ---------------------------------------------------------------- kernel 1
// Split fp32 rows into bf16 hi|lo (row-major [8192][512]) + row squared norms.
__global__ __launch_bounds__(256) void split_kernel(const float* __restrict__ src,
                                                    const float* __restrict__ tgt,
                                                    u16* __restrict__ Xs,
                                                    float* __restrict__ sq) {
    const int row = blockIdx.x;
    const int k   = threadIdx.x;                // 256 threads == DIM
    const float* p = (row < NS) ? (src + (size_t)row * DIM)
                                : (tgt + (size_t)(row - NS) * DIM);
    float x  = p[k];
    u16 hi   = f32_to_bf16(x);
    float lo = x - bf16_to_f32(hi);
    Xs[(size_t)row * KSPL + k]        = hi;
    Xs[(size_t)row * KSPL + DIM + k]  = f32_to_bf16(lo);

    __shared__ float red[256];
    red[k] = x * x;
    __syncthreads();
    for (int s = 128; s > 0; s >>= 1) {
        if (k < s) red[k] += red[k + s];
        __syncthreads();
    }
    if (k == 0) sq[row] = red[0];
}

// ---------------------------------------------------------------- kernel 2
// Column sums of the concatenated feature matrix (for closed-form bandwidth).
__global__ __launch_bounds__(256) void colsum_kernel(const float* __restrict__ src,
                                                     const float* __restrict__ tgt,
                                                     float* __restrict__ colsum) {
    const int k = threadIdx.x;
    float s = 0.f;
    for (int i = 0; i < NS; ++i) s += src[(size_t)i * DIM + k];
    for (int i = 0; i < NS; ++i) s += tgt[(size_t)i * DIM + k];
    colsum[k] = s;
}

// ---------------------------------------------------------------- kernel 3
// bandwidth -> 5 reciprocal scales; init argmin slots.
__global__ __launch_bounds__(256) void prep_kernel(const float* __restrict__ sq,
                                                   const float* __restrict__ colsum,
                                                   float* __restrict__ recips,
                                                   u64* __restrict__ rowSlots,
                                                   u64* __restrict__ colSlots) {
    const int tid = threadIdx.x;
    for (int i = tid; i < NS; i += 256) { rowSlots[i] = ~0ULL; colSlots[i] = ~0ULL; }

    float sp = 0.f;
    for (int i = tid; i < NTOT; i += 256) sp += sq[i];
    float c = colsum[tid];

    __shared__ float r1[256], r2[256];
    r1[tid] = sp; r2[tid] = c * c;
    __syncthreads();
    for (int s = 128; s > 0; s >>= 1) {
        if (tid < s) { r1[tid] += r1[tid + s]; r2[tid] += r2[tid + s]; }
        __syncthreads();
    }
    if (tid == 0) {
        // sum(l2) = 2*n*sum(sq) - 2*||colsum||^2
        float bwsum = 2.f * (float)NTOT * r1[0] - 2.f * r2[0];
        float n     = (float)NTOT;
        float bw    = bwsum / (n * n - n) / 4.f;          // / KERNEL_MUL^(NUM//2)
        for (int t = 0; t < 5; ++t)
            recips[t] = 1.f / (bw * (float)(1 << t));      // denom_i = bw * 2^i
    }
}

// ---------------------------------------------------------------- kernel 4
// Fused Gram GEMM (bf16 WMMA, split-K) + MK-MMD kernel sums + S-T argmins.
__global__ __launch_bounds__(256) void gram_kernel(const u16* __restrict__ Xs,
                                                   const float* __restrict__ sq,
                                                   const float* __restrict__ recips,
                                                   float* __restrict__ partials,
                                                   u64* __restrict__ rowSlots,
                                                   u64* __restrict__ colSlots) {
    __shared__ u16  As[BM * LDA];
    __shared__ u16  Bs[BN * LDA];
    __shared__ float sqRow[BM];
    __shared__ float sqCol[BN];
    __shared__ u64  rMin[BM];
    __shared__ u64  cMin[BN];
    __shared__ float red[256];

    const int tid  = threadIdx.x;
    const int lane = tid & 31;
    const int wid  = tid >> 5;
    const int wm   = wid & 3;          // 4 waves along M (32 rows each)
    const int wn   = wid >> 2;         // 2 waves along N (64 cols each)
    const int iBase = blockIdx.y * BM;
    const int jBase = blockIdx.x * BN;
    const bool isXY = (blockIdx.y < 32) && (blockIdx.x >= 32);

    if (tid < BM) { sqRow[tid] = sq[iBase + tid]; rMin[tid] = ~0ULL; }
    else          { sqCol[tid - BM] = sq[jBase + tid - BM]; cMin[tid - BM] = ~0ULL; }

    const float r0 = recips[0], r1 = recips[1], r2 = recips[2],
                r3 = recips[3], r4 = recips[4];

    v8f acc[2][4];
    #pragma unroll
    for (int a = 0; a < 2; ++a)
        #pragma unroll
        for (int b = 0; b < 4; ++b) acc[a][b] = (v8f){};

    // global->LDS staging: thread loads one 16-element (32B) chunk per tile
    const int ldRow  = tid >> 1;
    const int ldHalf = tid & 1;
    const u16* gA = Xs + (size_t)(iBase + ldRow) * KSPL + ldHalf * 16;
    const u16* gB = Xs + (size_t)(jBase + ldRow) * KSPL + ldHalf * 16;
    u16* sA = As + ldRow * LDA + ldHalf * 16;
    u16* sB = Bs + ldRow * LDA + ldHalf * 16;

    union Frag { v16bf v; uint4 q[2]; };
    const int r = lane & 15;
    const int h = lane >> 4;

    for (int kk = 0; kk < KSPL; kk += BK) {
        __syncthreads();
        *(uint4*)(sA)     = *(const uint4*)(gA + kk);
        *(uint4*)(sA + 8) = *(const uint4*)(gA + kk + 8);
        *(uint4*)(sB)     = *(const uint4*)(gB + kk);
        *(uint4*)(sB + 8) = *(const uint4*)(gB + kk + 8);
        if (kk + BK < KSPL) {
            __builtin_prefetch(gA + kk + BK, 0, 3);     // global_prefetch_b8
            __builtin_prefetch(gB + kk + BK, 0, 3);
        }
        __syncthreads();

        Frag fa[2], fb[4];
        #pragma unroll
        for (int ti = 0; ti < 2; ++ti) {
            // A 16x32 bf16 lane layout (ISA 7.12.2): lanes 0-15 K{0..7,16..23}, 16-31 K{8..15,24..31}
            const u16* pa = As + (wm * 32 + ti * 16 + r) * LDA;
            fa[ti].q[0] = *(const uint4*)(pa + h * 8);
            fa[ti].q[1] = *(const uint4*)(pa + 16 + h * 8);
        }
        #pragma unroll
        for (int tj = 0; tj < 4; ++tj) {
            // B 32x16 bf16 lane layout: lane = N col; lanes 0-15 K{0..7,16..23}, 16-31 K{8..15,24..31}
            const u16* pb = Bs + (wn * 64 + tj * 16 + r) * LDA;
            fb[tj].q[0] = *(const uint4*)(pb + h * 8);
            fb[tj].q[1] = *(const uint4*)(pb + 16 + h * 8);
        }
        #pragma unroll
        for (int ti = 0; ti < 2; ++ti)
            #pragma unroll
            for (int tj = 0; tj < 4; ++tj)
                acc[ti][tj] = __builtin_amdgcn_wmma_f32_16x16x32_bf16(
                    false, fa[ti].v, false, fb[tj].v,
                    (short)0, acc[ti][tj], false, false);
    }

    // ---- fused epilogue: 5-kernel Gaussian sums + packed argmin ----
    float locSum = 0.f;
    const int halfSel = (lane >> 4) ? 8 : 0;
    const int n16     = lane & 15;
    #pragma unroll
    for (int ti = 0; ti < 2; ++ti) {
        const int rowB = wm * 32 + ti * 16;
        #pragma unroll
        for (int tj = 0; tj < 4; ++tj) {
            const int colB = wn * 64 + tj * 16 + n16;
            const float sc = sqCol[colB];
            #pragma unroll
            for (int v = 0; v < 8; ++v) {
                const int   mi = rowB + v + halfSel;            // 0..127 in tile
                const float g  = acc[ti][tj][v];
                const float l2 = sqRow[mi] + sc - 2.f * g;
                locSum += __expf(-l2 * r0) + __expf(-l2 * r1) + __expf(-l2 * r2)
                        + __expf(-l2 * r3) + __expf(-l2 * r4);
                if (isXY) {
                    const float d2 = fmaxf(l2, 0.f);
                    const u64 key = ((u64)__float_as_uint(d2)) << 32;
                    atomicMin(&rMin[mi],  key | (u32)(jBase + colB - NS));
                    atomicMin(&cMin[colB], key | (u32)(iBase + mi));
                }
            }
        }
    }

    red[tid] = locSum;
    __syncthreads();
    if (tid == 0) {
        float s = 0.f;
        for (int t = 0; t < 256; ++t) s += red[t];
        partials[blockIdx.y * 64 + blockIdx.x] = s;
    }
    if (isXY) {
        if (tid < BM) atomicMin(&rowSlots[iBase + tid], rMin[tid]);
        else          atomicMin(&colSlots[jBase + tid - BM - NS], cMin[tid - BM]);
    }
}

// ---------------------------------------------------------------- kernel 5
// Cross-entropy + two distill losses (C=6) + deterministic transfer reduce.
__global__ __launch_bounds__(256) void finish_kernel(const float* __restrict__ sclf,
                                                     const float* __restrict__ tclf,
                                                     const int*  __restrict__ slab,
                                                     const u64*  __restrict__ rowSlots,
                                                     const u64*  __restrict__ colSlots,
                                                     const float* __restrict__ partials,
                                                     float* __restrict__ out) {
    const int tid = threadIdx.x;
    float accClf = 0.f, accDis = 0.f, accTr = 0.f;

    for (int i = tid; i < NS; i += 256) {
        // -- clf loss + distill #1 (teacher = source row i, student = target[min_idx[i]])
        const float* tch = sclf + (size_t)i * 6;
        float mt = tch[0];
        #pragma unroll
        for (int c = 1; c < 6; ++c) mt = fmaxf(mt, tch[c]);
        float et[6], st_ = 0.f;
        #pragma unroll
        for (int c = 0; c < 6; ++c) { et[c] = __expf(tch[c] - mt); st_ += et[c]; }
        float lseT = __logf(st_) + mt;
        accClf += lseT - tch[slab[i]];

        const int mi = (int)(rowSlots[i] & 0xFFFFFFFFULL);
        const float* stu = tclf + (size_t)mi * 6;
        float ms = stu[0];
        #pragma unroll
        for (int c = 1; c < 6; ++c) ms = fmaxf(ms, stu[c]);
        float ss = 0.f;
        #pragma unroll
        for (int c = 0; c < 6; ++c) ss += __expf(stu[c] - ms);
        float lseS = __logf(ss) + ms;
        #pragma unroll
        for (int c = 0; c < 6; ++c) accDis += (et[c] / st_) * (lseS - stu[c]);

        // -- distill #2 (teacher = source[min_idx_t[j]], student = target row j), j = i
        const int mj = (int)(colSlots[i] & 0xFFFFFFFFULL);
        const float* tch2 = sclf + (size_t)mj * 6;
        const float* stu2 = tclf + (size_t)i * 6;
        float mt2 = tch2[0], ms2 = stu2[0];
        #pragma unroll
        for (int c = 1; c < 6; ++c) { mt2 = fmaxf(mt2, tch2[c]); ms2 = fmaxf(ms2, stu2[c]); }
        float et2[6], st2 = 0.f, ss2 = 0.f;
        #pragma unroll
        for (int c = 0; c < 6; ++c) {
            et2[c] = __expf(tch2[c] - mt2); st2 += et2[c];
            ss2   += __expf(stu2[c] - ms2);
        }
        float lseS2 = __logf(ss2) + ms2;
        #pragma unroll
        for (int c = 0; c < 6; ++c) accDis += (et2[c] / st2) * (lseS2 - stu2[c]);
    }

    for (int p = tid; p < 4096; p += 256) {
        const int x = p & 63, y = p >> 6;
        const float sgn = ((y < 32) == (x < 32)) ? 1.f : -1.f;   // xx,yy: +  xy,yx: -
        accTr += sgn * partials[p];
    }

    __shared__ float rA[256], rB[256], rC[256];
    rA[tid] = accClf; rB[tid] = accDis; rC[tid] = accTr;
    __syncthreads();
    for (int s = 128; s > 0; s >>= 1) {
        if (tid < s) { rA[tid] += rA[tid + s]; rB[tid] += rB[tid + s]; rC[tid] += rC[tid + s]; }
        __syncthreads();
    }
    if (tid == 0) {
        out[0] = rA[0] / (float)NS;
        out[1] = rB[0] / (float)NS;
        out[2] = rC[0] / ((float)NS * (float)NS);
    }
}

// ---------------------------------------------------------------- launcher
extern "C" void kernel_launch(void* const* d_in, const int* in_sizes, int n_in,
                              void* d_out, int out_size, void* d_ws, size_t ws_size,
                              hipStream_t stream) {
    const float* sf   = (const float*)d_in[0];
    const float* tf   = (const float*)d_in[1];
    const float* sclf = (const float*)d_in[2];
    const float* tclf = (const float*)d_in[3];
    const int*   slab = (const int*)d_in[4];
    float* out = (float*)d_out;

    char* ws = (char*)d_ws;
    // workspace layout (all 256B aligned), ~8.5 MB total
    u16*   Xs       = (u16*)(ws);                                   // 8192*512*2 = 8388608
    float* sq       = (float*)(ws + 8388608);                       // 32768
    float* colsum   = (float*)(ws + 8388608 + 32768);               // 1024
    float* recips   = (float*)(ws + 8388608 + 32768 + 1024);        // 256
    u64*   rowSlots = (u64*)  (ws + 8388608 + 32768 + 1024 + 256);  // 32768
    u64*   colSlots = (u64*)  (ws + 8388608 + 32768 + 1024 + 256 + 32768);
    float* partials = (float*)(ws + 8388608 + 32768 + 1024 + 256 + 65536);

    split_kernel <<<NTOT, 256, 0, stream>>>(sf, tf, Xs, sq);
    colsum_kernel<<<1,    256, 0, stream>>>(sf, tf, colsum);
    prep_kernel  <<<1,    256, 0, stream>>>(sq, colsum, recips, rowSlots, colSlots);
    gram_kernel  <<<dim3(64, 64), 256, 0, stream>>>(Xs, sq, recips, partials, rowSlots, colSlots);
    finish_kernel<<<1,    256, 0, stream>>>(sclf, tclf, slab, rowSlots, colSlots, partials, out);
}
